// SSM_68728066670976
// MI455X (gfx1250) — matrix operand order repaired and checked by reading, not compiled, not run on hardware
//
#include <hip/hip_runtime.h>
#include <hip/hip_bf16.h>

// Problem constants (from reference)
#define DIMX      2048
#define DT_RANKC  64
#define DIM_INNER 2048
#define D_STATE   32
#define B_SZC     4
#define LSEQ      2048
#define MROWS     (B_SZC * LSEQ)          // 8192
#define NBC       (DT_RANKC + 2 * D_STATE) // 128

typedef __attribute__((ext_vector_type(16))) __bf16 v16bf;
typedef __attribute__((ext_vector_type(8)))  float  v8f;

// ---------------------------------------------------------------------------
// f32 -> bf16 conversion for the (small) weight matrices
// ---------------------------------------------------------------------------
__global__ void cvt_f32_bf16(const float* __restrict__ src,
                             __bf16* __restrict__ dst, int n) {
    int i = blockIdx.x * blockDim.x + threadIdx.x;
    if (i < n) dst[i] = (__bf16)src[i];
}

// ---------------------------------------------------------------------------
// GEMM1: deltaBC = x (8192x2048, f32->bf16 on the fly) @ W_bc^T (128x2048 bf16)
// One wave computes a 16(M) x 64(N) tile: 4 WMMAs per K-step reusing A frag.
// Epilogue: cols [0,64) -> dt_r as bf16 ; cols [64,128) -> B|C as f32.
// ---------------------------------------------------------------------------
__global__ void gemm1_deltaBC(const float*  __restrict__ x,
                              const __bf16* __restrict__ wbc,
                              __bf16*       __restrict__ dtr,
                              float*        __restrict__ bc) {
    const int lane  = threadIdx.x & 31;
    const int wave  = blockIdx.x * (blockDim.x >> 5) + (threadIdx.x >> 5);
    const int nt64  = wave & 1;        // which 64-wide N half of the 128 cols
    const int mt    = wave >> 1;       // 0..511 (M tile)
    const int row   = mt * 16 + (lane & 15);
    const int abase = (lane >> 4) * 16;    // A-frag K sub-base per lane half
    const int bbase = (lane >> 4) * 16;    // B-frag K sub-base per lane half
    const int ntbase = nt64 * 64;

    v8f acc[4] = {v8f{}, v8f{}, v8f{}, v8f{}};

    for (int k0 = 0; k0 < DIMX; k0 += 32) {
        // A fragment (16x32 bf16): one contiguous run of 16, cvt f32->bf16
        const float* pa = x + (size_t)row * DIMX + k0 + abase;
        float4 f0 = *(const float4*)(pa);
        float4 f1 = *(const float4*)(pa + 4);
        float4 f2 = *(const float4*)(pa + 8);
        float4 f3 = *(const float4*)(pa + 12);
        v16bf a;
        __bf16* ap = (__bf16*)&a;
        ap[0]=(__bf16)f0.x; ap[1]=(__bf16)f0.y; ap[2]=(__bf16)f0.z; ap[3]=(__bf16)f0.w;
        ap[4]=(__bf16)f1.x; ap[5]=(__bf16)f1.y; ap[6]=(__bf16)f1.z; ap[7]=(__bf16)f1.w;
        ap[8]=(__bf16)f2.x; ap[9]=(__bf16)f2.y; ap[10]=(__bf16)f2.z; ap[11]=(__bf16)f2.w;
        ap[12]=(__bf16)f3.x; ap[13]=(__bf16)f3.y; ap[14]=(__bf16)f3.z; ap[15]=(__bf16)f3.w;

        #pragma unroll
        for (int j = 0; j < 4; ++j) {
            // B fragment (32x16 bf16): one contiguous run of 16 from W_bc row
            const __bf16* pb = wbc +
                (size_t)(ntbase + j * 16 + (lane & 15)) * DIMX + k0 + bbase;
            v16bf b = *(const v16bf*)pb;
            acc[j] = __builtin_amdgcn_wmma_f32_16x16x32_bf16(
                false, a, false, b, (short)0, acc[j], false, false);
        }
    }

    #pragma unroll
    for (int j = 0; j < 4; ++j) {
        const float* cv = (const float*)&acc[j];
        #pragma unroll
        for (int r = 0; r < 8; ++r) {
            int m = mt * 16 + (lane >> 4) * 8 + r;
            int n = ntbase + j * 16 + (lane & 15);
            float v = cv[r];
            if (nt64 == 0) dtr[(size_t)m * DT_RANKC + n] = (__bf16)v;
            else           bc [(size_t)m * 64 + (n - 64)] = v;
        }
    }
}

// ---------------------------------------------------------------------------
// GEMM2: dt = dt_r (8192x64 bf16) @ W_dt^T (2048x64 bf16); fused softplus.
// One wave computes a 16(M) x 64(N) tile; K loop fully unrolled (2 steps).
// ---------------------------------------------------------------------------
__global__ void gemm2_delta(const __bf16* __restrict__ dtr,
                            const __bf16* __restrict__ wdt,
                            const float*  __restrict__ b_dt,
                            float*        __restrict__ delta) {
    const int lane  = threadIdx.x & 31;
    const int wave  = blockIdx.x * (blockDim.x >> 5) + (threadIdx.x >> 5);
    const int ntile = wave & 31;       // 32 column groups of 64
    const int mt    = wave >> 5;       // 0..511
    const int row   = mt * 16 + (lane & 15);
    const int abase = (lane >> 4) * 16;
    const int bbase = (lane >> 4) * 16;
    const int ntbase = ntile * 64;

    v8f acc[4] = {v8f{}, v8f{}, v8f{}, v8f{}};

    #pragma unroll
    for (int k0 = 0; k0 < DT_RANKC; k0 += 32) {
        const __bf16* pa = dtr + (size_t)row * DT_RANKC + k0 + abase;
        v16bf a;
        *((uint4*)&a + 0) = *(const uint4*)(pa);        // K runs base+0..7
        *((uint4*)&a + 1) = *(const uint4*)(pa + 8);    // K runs base+8..15
        #pragma unroll
        for (int j = 0; j < 4; ++j) {
            const __bf16* pb = wdt +
                (size_t)(ntbase + j * 16 + (lane & 15)) * DT_RANKC + k0 + bbase;
            v16bf b = *(const v16bf*)pb;
            acc[j] = __builtin_amdgcn_wmma_f32_16x16x32_bf16(
                false, a, false, b, (short)0, acc[j], false, false);
        }
    }

    #pragma unroll
    for (int j = 0; j < 4; ++j) {
        const float* cv = (const float*)&acc[j];
        #pragma unroll
        for (int r = 0; r < 8; ++r) {
            int m = mt * 16 + (lane >> 4) * 8 + r;
            int n = ntbase + j * 16 + (lane & 15);
            float v = cv[r] + b_dt[n];
            float sp = (v > 20.0f) ? v : __logf(1.0f + __expf(v));
            delta[(size_t)m * DIM_INNER + n] = sp;
        }
    }
}

// ---------------------------------------------------------------------------
// Selective scan: 4 lanes per (b,d) channel, 8 states per lane.
// h[n] = exp(delta*A[n])*h[n] + delta*u*B[n];  y = sum_n h[n]*C[n] + u*D
// Reduction over the 4-lane group via two shfl_xor hops.
// ---------------------------------------------------------------------------
__global__ void scan_kernel(const float* __restrict__ x,
                            const float* __restrict__ delta,
                            const float* __restrict__ bc,
                            const float* __restrict__ A_log,
                            const float* __restrict__ D_param,
                            float*       __restrict__ y) {
    const int gid  = blockIdx.x * blockDim.x + threadIdx.x;
    const int chan = gid >> 2;          // 0..8191 = b*DIM_INNER + d
    const int sub  = gid & 3;           // which 8-state slice of n
    const int b    = chan / DIM_INNER;
    const int d    = chan - b * DIM_INNER;

    // A = -exp(A_log); pre-scale by log2(e) so the loop uses v_exp_f32 (base-2).
    float a2[8];
    #pragma unroll
    for (int j = 0; j < 8; ++j)
        a2[j] = -__expf(A_log[d * D_STATE + sub * 8 + j]) * 1.44269504088896340736f;
    const float Dp = D_param[d];

    float h[8];
    #pragma unroll
    for (int j = 0; j < 8; ++j) h[j] = 0.0f;

    for (int t = 0; t < LSEQ; ++t) {
        const size_t m = (size_t)b * LSEQ + t;
        float dlt = delta[m * DIM_INNER + d];
        float u   = x    [m * DIM_INNER + d];
        float du  = dlt * u;

        const float* pB = bc + m * 64 + sub * 8;   // B = cols [0,32)
        const float* pC = pB + 32;                 // C = cols [32,64)
        float4 b0 = *(const float4*)(pB);
        float4 b1 = *(const float4*)(pB + 4);
        float4 c0 = *(const float4*)(pC);
        float4 c1 = *(const float4*)(pC + 4);
        float bv[8] = {b0.x,b0.y,b0.z,b0.w,b1.x,b1.y,b1.z,b1.w};
        float cv[8] = {c0.x,c0.y,c0.z,c0.w,c1.x,c1.y,c1.z,c1.w};

        float acc = 0.0f;
        #pragma unroll
        for (int j = 0; j < 8; ++j) {
            float e = __builtin_amdgcn_exp2f(dlt * a2[j]);  // exp(delta * A[n])
            h[j] = fmaf(e, h[j], du * bv[j]);
            acc  = fmaf(h[j], cv[j], acc);
        }
        acc += __shfl_xor(acc, 1, 32);
        acc += __shfl_xor(acc, 2, 32);
        if (sub == 0) y[m * DIM_INNER + d] = fmaf(u, Dp, acc);
    }
}

// ---------------------------------------------------------------------------
extern "C" void kernel_launch(void* const* d_in, const int* in_sizes, int n_in,
                              void* d_out, int out_size, void* d_ws, size_t ws_size,
                              hipStream_t stream) {
    const float* x       = (const float*)d_in[0];
    // d_in[1] = z : unused by the reference
    const float* W_bc    = (const float*)d_in[2];
    const float* W_dt    = (const float*)d_in[3];
    const float* b_dt    = (const float*)d_in[4];
    const float* A_log   = (const float*)d_in[5];
    const float* D_param = (const float*)d_in[6];
    float* y = (float*)d_out;

    // Workspace carve-up (all offsets 256B-aligned); total ~68 MB.
    char* ws = (char*)d_ws;
    size_t o = 0;
    __bf16* wbc_bf = (__bf16*)(ws + o); o += (size_t)NBC * DIMX * 2;         // 512 KB
    __bf16* wdt_bf = (__bf16*)(ws + o); o += (size_t)DIM_INNER * DT_RANKC*2; // 256 KB
    __bf16* dtr_bf = (__bf16*)(ws + o); o += (size_t)MROWS * DT_RANKC * 2;   // 1 MB
    float*  bc     = (float*) (ws + o); o += (size_t)MROWS * 64 * 4;         // 2 MB
    float*  delta  = (float*) (ws + o);                                      // 64 MB
    (void)ws_size; (void)in_sizes; (void)n_in; (void)out_size;

    // 1) weight conversions f32 -> bf16
    cvt_f32_bf16<<<(NBC * DIMX + 255) / 256, 256, 0, stream>>>(W_bc, wbc_bf, NBC * DIMX);
    cvt_f32_bf16<<<(DIM_INNER * DT_RANKC + 255) / 256, 256, 0, stream>>>(W_dt, wdt_bf, DIM_INNER * DT_RANKC);

    // 2) GEMM1: 1024 wave-jobs (512 M-tiles x 2 N-halves), 4 waves/block
    gemm1_deltaBC<<<256, 128, 0, stream>>>(x, wbc_bf, dtr_bf, bc);

    // 3) GEMM2 + softplus: 16384 wave-jobs (512 x 32), 4 waves/block
    gemm2_delta<<<4096, 128, 0, stream>>>(dtr_bf, wdt_bf, b_dt, delta);

    // 4) selective scan: 8192 channels x 4 lanes = 32768 threads
    scan_kernel<<<128, 256, 0, stream>>>(x, delta, bc, A_log, D_param, y);
}
